// JointGNN_34127810134111
// MI455X (gfx1250) — compile-verified, matrix operand order checked by reading, not run on hardware
//
#include <hip/hip_runtime.h>
#include <hip/hip_bf16.h>
#include <cmath>

// ---------------------------------------------------------------------------
// CDNA5 (gfx1250) WMMA types
// ---------------------------------------------------------------------------
typedef __attribute__((ext_vector_type(16))) __bf16 v16bf;
typedef __attribute__((ext_vector_type(8)))  float  v8f;

#define DEV __device__ __forceinline__

DEV __bf16 f2bf(float f) {
#if defined(__gfx1250__)
  return (__bf16)f;                 // native v_cvt_pk_bf16_f32 path on CDNA5
#else
  unsigned u = __builtin_bit_cast(unsigned, f);
  u = u + 0x7fffu + ((u >> 16) & 1u);
  unsigned short s = (unsigned short)(u >> 16);
  return __builtin_bit_cast(__bf16, s);
#endif
}

DEV float sigmoidf_(float x) { return 1.0f / (1.0f + __expf(-x)); }

// A tile: 16x32 (MxK) bf16, ISA 7.12.2 layout. rowp = base of this lane's row
// (m = lane&15). Two contiguous 32B runs -> 4x global_load_b128.
DEV v16bf load_a(const float* __restrict__ rowp, int half) {
  const float4* p0 = (const float4*)(rowp + (half << 3));
  const float4* p1 = (const float4*)(rowp + 16 + (half << 3));
  const float4 x0 = p0[0], x1 = p0[1];
  const float4 y0 = p1[0], y1 = p1[1];
  v16bf a;
  a[0]  = f2bf(x0.x); a[1]  = f2bf(x0.y); a[2]  = f2bf(x0.z); a[3]  = f2bf(x0.w);
  a[4]  = f2bf(x1.x); a[5]  = f2bf(x1.y); a[6]  = f2bf(x1.z); a[7]  = f2bf(x1.w);
  a[8]  = f2bf(y0.x); a[9]  = f2bf(y0.y); a[10] = f2bf(y0.z); a[11] = f2bf(y0.w);
  a[12] = f2bf(y1.x); a[13] = f2bf(y1.y); a[14] = f2bf(y1.z); a[15] = f2bf(y1.w);
  return a;
}

// ---------------------------------------------------------------------------
// Pack fp32 weight [K,Nc] into bf16 WMMA B-fragments, layout [nt][kt][lane].
// One wave per 32x16 (KxN) tile; each lane stores its v16bf fragment (32B).
// ---------------------------------------------------------------------------
__global__ void __launch_bounds__(256)
pack_w_bf16(const float* __restrict__ W, __bf16* __restrict__ out, int K, int Nc)
{
  const int lane = threadIdx.x & 31;
  const int wid  = threadIdx.x >> 5;
  const int ktiles = K >> 5, ntiles = Nc >> 4;
  const int tile = (int)blockIdx.x * 8 + wid;
  if (tile >= ktiles * ntiles) return;
  const int nt = tile / ktiles, kt = tile % ktiles;
  const int half = lane >> 4;
  const int n = (nt << 4) + (lane & 15);
  v16bf frag;
#pragma unroll
  for (int j = 0; j < 8; ++j) {
    const int k = (kt << 5) + ((j >> 2) << 4) + (half << 3) + ((j & 3) << 1);
    frag[2 * j]     = f2bf(W[(size_t)k * Nc + n]);
    frag[2 * j + 1] = f2bf(W[(size_t)(k + 1) * Nc + n]);
  }
  ((v16bf*)out)[(size_t)tile * 32 + lane] = frag;
}

// ---------------------------------------------------------------------------
// Generic WMMA GEMM: C[M,Nc] = act(A @ W + bias), W pre-packed bf16 fragments.
// A fp32, optionally 3 column-segments of width 128 with row-gather indices.
// One wave computes NT adjacent 16x16 tiles (A fragment reused NT times).
// ---------------------------------------------------------------------------
template <int NT>
__global__ void __launch_bounds__(256)
gemm_bias_act(const float* __restrict__ A0, const int* __restrict__ I0,
              const float* __restrict__ A1, const int* __restrict__ I1,
              const float* __restrict__ A2, const int* __restrict__ I2,
              int nseg, int M, int K, int Nc,
              const __bf16* __restrict__ Wp, const float* __restrict__ bias,
              float* __restrict__ C, int act)
{
  const int lane = threadIdx.x & 31;
  const int wid  = threadIdx.x >> 5;
  const int ngroups = Nc / (16 * NT);
  const int ktiles  = K >> 5;
  const long tid   = (long)blockIdx.x * 8 + wid;
  const long total = (long)(M >> 4) * ngroups;
  if (tid >= total) return;
  const int mt = (int)(tid / ngroups);
  const int ng = (int)(tid % ngroups);
  const int row0 = mt << 4;
  const int n0   = ng * (16 * NT);
  const int half = lane >> 4;

  v8f acc[NT] = {};
  for (int seg = 0; seg < nseg; ++seg) {
    const float* ap; const int* ip; int ld, kw, kbase;
    if (nseg == 1) { ap = A0; ip = I0; ld = K; kw = K; kbase = 0; }
    else {
      ap = (seg == 0) ? A0 : ((seg == 1) ? A1 : A2);
      ip = (seg == 0) ? I0 : ((seg == 1) ? I1 : I2);
      ld = 128; kw = 128; kbase = seg << 7;
    }
    int r = row0 + (lane & 15);
    if (ip) r = ip[r];                       // gather: resolved once per segment
    const float* arow = ap + (size_t)r * ld;
    __builtin_prefetch(arow, 0, 3);          // global_prefetch for gathered row
    for (int kk = 0; kk < kw; kk += 32) {
      const v16bf a = load_a(arow + kk, half);
      const int kt = (kbase + kk) >> 5;
#pragma unroll
      for (int t = 0; t < NT; ++t) {
        const v16bf b =
            ((const v16bf*)Wp)[((size_t)(ng * NT + t) * ktiles + kt) * 32 + lane];
        acc[t] = __builtin_amdgcn_wmma_f32_16x16x32_bf16(false, a, false, b,
                                                         (short)0, acc[t],
                                                         false, false);
      }
    }
  }
#pragma unroll
  for (int t = 0; t < NT; ++t) {
    const int n = n0 + (t << 4) + (lane & 15);
    const float bn = bias[n];
#pragma unroll
    for (int j = 0; j < 8; ++j) {
      const int m = row0 + j + (half << 3);
      float v = acc[t][j] + bn;
      if (act == 1) v = fmaxf(v, 0.0f);
      C[(size_t)m * Nc + n] = v;
    }
  }
}

// ---------------------------------------------------------------------------
// Fused GRUCell: out = (1-z)*tanh(gi_n + r*gh_n) + z*h
//   gi = X @ Wih + bih, gh = H @ Whh + bhh  (HASH=false -> h=0)
// Wih/Whh pre-packed bf16 fragments [nt=24][kt=4][lane]. One wave / 16 rows.
// Compile-time HASH specialization -> branch-free dual-WMMA inner loop.
// ---------------------------------------------------------------------------
template <bool HASH>
__global__ void __launch_bounds__(128)
gru_fused(const float* __restrict__ X, const float* __restrict__ H,
          const __bf16* __restrict__ Wihp, const __bf16* __restrict__ Whhp,
          const float* __restrict__ bih, const float* __restrict__ bhh,
          float* __restrict__ OUT, int M)
{
  __shared__ __attribute__((aligned(16))) float s[4][3][16][128]; // r/z/n, 96KB
  const int lane = threadIdx.x & 31;
  const int wid  = threadIdx.x >> 5;
  const int row0 = (((int)blockIdx.x << 2) + wid) << 4;
  if (row0 >= M) return;
  const int half = lane >> 4;
  const int nl   = lane & 15;

  const float* xrow = X + (size_t)(row0 + nl) * 128;
  const float* hrow = HASH ? H + (size_t)(row0 + nl) * 128 : nullptr;

  for (int nt = 0; nt < 24; ++nt) {
    v8f ai = {}; v8f ah = {};
#pragma unroll
    for (int kt = 0; kt < 4; ++kt) {
      const v16bf ax = load_a(xrow + (kt << 5), half);
      const v16bf bi = ((const v16bf*)Wihp)[(size_t)(nt * 4 + kt) * 32 + lane];
      ai = __builtin_amdgcn_wmma_f32_16x16x32_bf16(false, ax, false, bi,
                                                   (short)0, ai, false, false);
      if (HASH) {
        const v16bf axh = load_a(hrow + (kt << 5), half);
        const v16bf bh  = ((const v16bf*)Whhp)[(size_t)(nt * 4 + kt) * 32 + lane];
        ah = __builtin_amdgcn_wmma_f32_16x16x32_bf16(false, axh, false, bh,
                                                     (short)0, ah, false, false);
      }
    }
    const int col    = (nt << 4) + nl;
    const int region = nt >> 3;                 // 0=r, 1=z, 2=n
    const int ccol   = col - (region << 7);
    const float bi_ = bih[col], bh_ = bhh[col];
#pragma unroll
    for (int j = 0; j < 8; ++j) {
      const int m = j + (half << 3);
      const float gi = ai[j] + bi_;
      const float gh = (HASH ? ah[j] : 0.0f) + bh_;
      if (region == 0) {
        s[wid][0][m][ccol] = sigmoidf_(gi + gh);
      } else if (region == 1) {
        s[wid][1][m][ccol] = sigmoidf_(gi + gh);
      } else {
        const float r = s[wid][0][m][ccol];     // written earlier, same wave
        s[wid][2][m][ccol] = tanhf(gi + r * gh);
      }
    }
  }
  // epilogue: read h before overwriting (safe for OUT==H, rows tile-exclusive)
  for (int t = lane; t < 512; t += 32) {        // 2048 floats as float4
    const int m = t >> 5, c4 = t & 31;
    const float4 z4 = ((const float4*)&s[wid][1][m][0])[c4];
    const float4 n4 = ((const float4*)&s[wid][2][m][0])[c4];
    float4 h4 = make_float4(0.f, 0.f, 0.f, 0.f);
    if (HASH) h4 = ((const float4*)(H + (size_t)(row0 + m) * 128))[c4];
    float4 o;
    o.x = (1.0f - z4.x) * n4.x + z4.x * h4.x;
    o.y = (1.0f - z4.y) * n4.y + z4.y * h4.y;
    o.z = (1.0f - z4.z) * n4.z + z4.z * h4.z;
    o.w = (1.0f - z4.w) * n4.w + z4.w * h4.w;
    ((float4*)(OUT + (size_t)(row0 + m) * 128))[c4] = o;
  }
}

// ---------------------------------------------------------------------------
// geo gating: node += sigmoid([node,geo] @ geo_w + geo_b) * sigmoid(geo)
// one wave32 per node; float4 loads; butterfly reduction for the scalar gate.
// ---------------------------------------------------------------------------
__global__ void __launch_bounds__(256)
geo_gate(float* __restrict__ node, const float* __restrict__ geo,
         const float* __restrict__ gw, const float* __restrict__ gb, int Nn)
{
  const int lane = threadIdx.x & 31;
  const int w = ((int)blockIdx.x << 3) + (threadIdx.x >> 5);
  if (w >= Nn) return;
  const float4 nd  = ((const float4*)(node + (size_t)w * 128))[lane];
  const float4 gg  = ((const float4*)(geo  + (size_t)w * 128))[lane];
  const float4 gw1 = ((const float4*)gw)[lane];
  const float4 gw2 = ((const float4*)(gw + 128))[lane];
  float part = nd.x * gw1.x + nd.y * gw1.y + nd.z * gw1.z + nd.w * gw1.w
             + gg.x * gw2.x + gg.y * gw2.y + gg.z * gw2.z + gg.w * gw2.w;
#pragma unroll
  for (int m = 16; m >= 1; m >>= 1) part += __shfl_xor(part, m, 32);
  const float gate = sigmoidf_(part + gb[0]);
  float4 o;
  o.x = nd.x + gate * sigmoidf_(gg.x);
  o.y = nd.y + gate * sigmoidf_(gg.y);
  o.z = nd.z + gate * sigmoidf_(gg.z);
  o.w = nd.w + gate * sigmoidf_(gg.w);
  ((float4*)(node + (size_t)w * 128))[lane] = o;
}

// ---------------------------------------------------------------------------
// FAN attention: 32 edges per block, thread = (edge, head).
// q/k/v restaged through LDS with coalesced float4 loads; probs staged in LDS
// and written back coalesced; prob*v scatter-added into agg (f32 atomics).
// ---------------------------------------------------------------------------
__global__ void __launch_bounds__(256)
fan_attention(const float* __restrict__ qb, const float* __restrict__ kb,
              const float* __restrict__ vb, const int* __restrict__ row,
              float* __restrict__ agg, float* __restrict__ probs,
              const float* __restrict__ Wa1, const float* __restrict__ ba1,
              const float* __restrict__ Wa2, const float* __restrict__ ba2,
              float inv_temp, int E)
{
  __shared__ float w1[32 * 32], w2[32 * 16], b1[32], b2[16];
  __shared__ __attribute__((aligned(16))) float sq[32 * 128];
  __shared__ __attribute__((aligned(16))) float sk[32 * 128];
  __shared__ __attribute__((aligned(16))) float sv[32 * 128];
  const int tid = threadIdx.x;
  for (int t = tid; t < 1024; t += 256) w1[t] = Wa1[t];
  for (int t = tid; t < 512;  t += 256) w2[t] = Wa2[t];
  if (tid < 32) b1[tid] = ba1[tid];
  if (tid < 16) b2[tid] = ba2[tid];

  const int e0 = (int)blockIdx.x * 32;
  int nvalid = E - e0; if (nvalid > 32) nvalid = 32;
  const int nf4 = nvalid * 32;                 // float4 count per array
  for (int t = tid; t < nf4; t += 256) {
    ((float4*)sq)[t] = ((const float4*)(qb + (size_t)e0 * 128))[t];
    ((float4*)sk)[t] = ((const float4*)(kb + (size_t)e0 * 128))[t];
    ((float4*)sv)[t] = ((const float4*)(vb + (size_t)e0 * 128))[t];
  }
  __syncthreads();

  const int el = tid >> 3;                     // local edge 0..31
  const int h  = tid & 7;
  const int e  = e0 + el;
  const bool active = (e < E);

  float p[16];
  if (active) {
    float a[32];
#pragma unroll
    for (int c = 0; c < 16; ++c) {
      a[c]      = sq[el * 128 + c * 8 + h];
      a[16 + c] = sk[el * 128 + c * 8 + h];
    }
    float h1[32];
#pragma unroll
    for (int o = 0; o < 32; ++o) {
      float sum = b1[o];
#pragma unroll
      for (int c = 0; c < 32; ++c) sum += a[c] * w1[c * 32 + o];
      h1[o] = fmaxf(sum, 0.0f);
    }
    float mx = -1e30f;
#pragma unroll
    for (int o = 0; o < 16; ++o) {
      float sum = b2[o];
#pragma unroll
      for (int c = 0; c < 32; ++c) sum += h1[c] * w2[c * 16 + o];
      p[o] = sum * inv_temp;
      mx = fmaxf(mx, p[o]);
    }
    float denom = 0.0f;
#pragma unroll
    for (int o = 0; o < 16; ++o) { p[o] = __expf(p[o] - mx); denom += p[o]; }
    const float rd = 1.0f / denom;
#pragma unroll
    for (int o = 0; o < 16; ++o) p[o] *= rd;
    const int r = row[e];
    // stage probs into sq (this edge's slice is only touched by its own wave)
#pragma unroll
    for (int o = 0; o < 16; ++o) sq[el * 128 + o * 8 + h] = p[o];
    // scatter-sum prob * v into agg
#pragma unroll
    for (int o = 0; o < 16; ++o)
      atomicAdd(&agg[(size_t)r * 128 + o * 8 + h],
                p[o] * sv[el * 128 + o * 8 + h]);
  }
  __syncthreads();
  for (int t = tid; t < nf4; t += 256)
    ((float4*)(probs + (size_t)e0 * 128))[t] = ((const float4*)sq)[t];
}

__global__ void fill_zero(float* __restrict__ p, long n) {
  const long i = (long)blockIdx.x * blockDim.x + threadIdx.x;
  if (i < n) p[i] = 0.0f;
}

// ---------------------------------------------------------------------------
// Orchestration
// ---------------------------------------------------------------------------
extern "C" void kernel_launch(void* const* d_in, const int* in_sizes, int n_in,
                              void* d_out, int out_size, void* d_ws, size_t ws_size,
                              hipStream_t stream) {
  (void)n_in; (void)out_size; (void)ws_size;
  const int D = 128;
  const int Nn = in_sizes[0] / D;        // 20000
  const int E  = in_sizes[1] / D;        // 320000
  const int L  = in_sizes[15] / (2 * D); // be1 is [L, 2D]
  const int dep = in_sizes[27] / L;      // ba2 is [L, dep] -> 16
  const float inv_temp = 1.0f / sqrtf((float)dep);

  const float* node_in = (const float*)d_in[0];
  const float* edge_in = (const float*)d_in[1];
  const float* geo     = (const float*)d_in[2];
  const int*   eidx    = (const int*)d_in[3];
  const int* row = eidx;
  const int* col = eidx + E;
  const float* gnw_ih = (const float*)d_in[4];
  const float* gnw_hh = (const float*)d_in[5];
  const float* gnb_ih = (const float*)d_in[6];
  const float* gnb_hh = (const float*)d_in[7];
  const float* gew_ih = (const float*)d_in[8];
  const float* gew_hh = (const float*)d_in[9];
  const float* geb_ih = (const float*)d_in[10];
  const float* geb_hh = (const float*)d_in[11];
  const float* geo_w  = (const float*)d_in[12];
  const float* geo_b  = (const float*)d_in[13];
  const float* We1 = (const float*)d_in[14];
  const float* be1 = (const float*)d_in[15];
  const float* We2 = (const float*)d_in[16];
  const float* be2 = (const float*)d_in[17];
  const float* Wq = (const float*)d_in[18];
  const float* bq = (const float*)d_in[19];
  const float* Wk = (const float*)d_in[20];
  const float* bk = (const float*)d_in[21];
  const float* Wv = (const float*)d_in[22];
  const float* bv = (const float*)d_in[23];
  const float* Wa1 = (const float*)d_in[24];
  const float* ba1 = (const float*)d_in[25];
  const float* Wa2 = (const float*)d_in[26];
  const float* ba2 = (const float*)d_in[27];
  const float* Wu1 = (const float*)d_in[28];
  const float* bu1 = (const float*)d_in[29];
  const float* Wu2 = (const float*)d_in[30];
  const float* bu2 = (const float*)d_in[31];

  float* out = (float*)d_out;
  float* node_out  = out;
  float* edge_out  = out + (size_t)Nn * D;
  float* probs_out = edge_out + (size_t)E * D;

  // workspace layout (fp32 units)
  float* w = (float*)d_ws;
  float* node_cur = w;  w += (size_t)Nn * D;
  float* edge_cur = w;  w += (size_t)E * D;
  float* edge_msg = w;  w += (size_t)E * D;
  float* big      = w;  w += (size_t)E * 3 * D;   // hid1 (E x 2D) then q|k|v
  float* hid1 = big;
  float* qb = big;
  float* kb = big + (size_t)E * D;
  float* vb = big + (size_t)E * 2 * D;
  float* agg      = w;  w += (size_t)Nn * D;
  float* nhid     = w;  w += (size_t)Nn * 2 * D;
  float* node_msg = w;  w += (size_t)Nn * D;
  // packed bf16 weight fragment buffers (counted in floats; 2 bf16 per float)
  __bf16* pk_gn_ih = (__bf16*)w;  w += (size_t)D * 3 * D / 2;
  __bf16* pk_gn_hh = (__bf16*)w;  w += (size_t)D * 3 * D / 2;
  __bf16* pk_ge_ih = (__bf16*)w;  w += (size_t)D * 3 * D / 2;
  __bf16* pk_ge_hh = (__bf16*)w;  w += (size_t)D * 3 * D / 2;
  __bf16* pk_gemm  = (__bf16*)w;  w += (size_t)3 * D * 2 * D / 2;

  auto pack = [&](const float* W, __bf16* dst, int K, int Nc) {
    const int tiles = (K / 32) * (Nc / 16);
    pack_w_bf16<<<(tiles + 7) / 8, 256, 0, stream>>>(W, dst, K, Nc);
  };
  auto gemm = [&](const float* A0, const int* I0, const float* A1, const int* I1,
                  const float* A2, const int* I2, int nseg,
                  int M, int K, int Nc, const float* W, const float* B,
                  float* C, int act) {
    pack(W, pk_gemm, K, Nc);
    const long total = (long)(M / 16) * (Nc / 32);   // NT = 2
    const int blocks = (int)((total + 7) / 8);
    gemm_bias_act<2><<<blocks, 256, 0, stream>>>(A0, I0, A1, I1, A2, I2,
                                                 nseg, M, K, Nc, pk_gemm, B,
                                                 C, act);
  };
  auto gru0 = [&](const float* X, const __bf16* Wihp, const __bf16* Whhp,
                  const float* bih, const float* bhh, float* O, int M) {
    gru_fused<false><<<((M / 16) + 3) / 4, 128, 0, stream>>>(
        X, nullptr, Wihp, Whhp, bih, bhh, O, M);
  };
  auto gru = [&](const float* X, const float* H, const __bf16* Wihp,
                 const __bf16* Whhp, const float* bih, const float* bhh,
                 float* O, int M) {
    gru_fused<true><<<((M / 16) + 3) / 4, 128, 0, stream>>>(
        X, H, Wihp, Whhp, bih, bhh, O, M);
  };

  // pack the 4 GRU weights once (reused by all GRU steps)
  pack(gnw_ih, pk_gn_ih, D, 3 * D);
  pack(gnw_hh, pk_gn_hh, D, 3 * D);
  pack(gew_ih, pk_ge_ih, D, 3 * D);
  pack(gew_hh, pk_ge_hh, D, 3 * D);

  // initial GRU step (h = 0)
  gru0(node_in, pk_gn_ih, pk_gn_hh, gnb_ih, gnb_hh, node_cur, Nn);
  gru0(edge_in, pk_ge_ih, pk_ge_hh, geb_ih, geb_hh, edge_cur, E);

  for (int i = 0; i < L; ++i) {
    const int lastAct = (i < L - 1 || L == 1) ? 1 : 0;

    // geo gating (in-place on node_cur)
    geo_gate<<<(Nn + 7) / 8, 256, 0, stream>>>(node_cur, geo, geo_w, geo_b, Nn);

    // triplet MLP: hid1 = relu([x_i, edge, x_j] @ We1 + be1); edge_msg = hid1 @ We2 + be2
    gemm(node_cur, row, edge_cur, nullptr, node_cur, col, 3,
         E, 3 * D, 2 * D, We1 + (size_t)i * 3 * D * 2 * D, be1 + (size_t)i * 2 * D,
         hid1, 1);
    gemm(hid1, nullptr, nullptr, nullptr, nullptr, nullptr, 1,
         E, 2 * D, D, We2 + (size_t)i * 2 * D * D, be2 + (size_t)i * D,
         edge_msg, lastAct);

    // q/k/v projections (overwrite the hid1 region — hid1 already consumed)
    gemm(node_cur, row, nullptr, nullptr, nullptr, nullptr, 1,
         E, D, D, Wq + (size_t)i * D * D, bq + (size_t)i * D, qb, 0);
    gemm(edge_cur, nullptr, nullptr, nullptr, nullptr, nullptr, 1,
         E, D, D, Wk + (size_t)i * D * D, bk + (size_t)i * D, kb, 0);
    gemm(node_cur, col, nullptr, nullptr, nullptr, nullptr, 1,
         E, D, D, Wv + (size_t)i * D * D, bv + (size_t)i * D, vb, 0);

    // attention + scatter-sum
    const long an = (long)Nn * D;
    fill_zero<<<(int)((an + 255) / 256), 256, 0, stream>>>(agg, an);
    fan_attention<<<(E + 31) / 32, 256, 0, stream>>>(
        qb, kb, vb, row, agg, probs_out + (size_t)i * E * 128,
        Wa1 + (size_t)i * 1024, ba1 + (size_t)i * 32,
        Wa2 + (size_t)i * 512,  ba2 + (size_t)i * 16, inv_temp, E);

    // node update MLP: nhid = relu([node, agg] @ Wu1 + bu1); node_msg = nhid @ Wu2 + bu2
    gemm(node_cur, nullptr, agg, nullptr, nullptr, nullptr, 2,
         Nn, 2 * D, 2 * D, Wu1 + (size_t)i * 2 * D * 2 * D, bu1 + (size_t)i * 2 * D,
         nhid, 1);
    gemm(nhid, nullptr, nullptr, nullptr, nullptr, nullptr, 1,
         Nn, 2 * D, D, Wu2 + (size_t)i * 2 * D * D, bu2 + (size_t)i * D,
         node_msg, lastAct);

    // GRU state updates (final layer writes straight into d_out)
    float* nout = (i == L - 1) ? node_out : node_cur;
    float* eout = (i == L - 1) ? edge_out : edge_cur;
    gru(node_msg, node_cur, pk_gn_ih, pk_gn_hh, gnb_ih, gnb_hh, nout, Nn);
    gru(edge_msg, edge_cur, pk_ge_ih, pk_ge_hh, geb_ih, geb_hh, eout, E);
  }
}
